// SizeBlock_11940009083488
// MI455X (gfx1250) — matrix operand order repaired and checked by reading, not compiled, not run on hardware
//
#include <hip/hip_runtime.h>
#include <math.h>

typedef __attribute__((ext_vector_type(16))) _Float16 v16h;
typedef __attribute__((ext_vector_type(8)))  _Float16 v8h;
typedef __attribute__((ext_vector_type(8)))  float    v8f;

#define WMMA_F16(a,b,c) __builtin_amdgcn_wmma_f32_16x16x32_f16(false,(a),false,(b),(short)0,(c),false,false)

// Fixed problem shape from the reference.
constexpr int BB = 8, HH = 128, WW = 128, OO = 64;

// ---------------------------------------------------------------------------
// Layout-convert: x NCHW f32 -> NHWC f16 (channels-last, WMMA-friendly)
// ---------------------------------------------------------------------------
__global__ void to_nhwc_f16(const float* __restrict__ x, _Float16* __restrict__ xo, int total) {
    int tid = blockIdx.x * blockDim.x + threadIdx.x;
    if (tid >= total) return;
    int c   = tid & 63;
    int pix = tid >> 6;
    int w = pix & 127, h = (pix >> 7) & 127, b = pix >> 14;
    xo[tid] = (_Float16)x[((b * 64 + c) * 128 + h) * 128 + w];
}

// ---------------------------------------------------------------------------
// Pack OIHW f32 weights into WMMA A-fragment layout (f16):
//   pk[((mt*KC + kc)*32 + lane)*16 + j] = W[m][k]
//   m = mt*16 + (lane&15), g = lane>>4, K = kc*32 + j + 16*g
//   k ordering: k = (ki*3+kj)*C + c
// ---------------------------------------------------------------------------
__global__ void pack_w(const float* __restrict__ w, _Float16* __restrict__ pk,
                       int O, int C, int total) {
    int tid = blockIdx.x * blockDim.x + threadIdx.x;
    if (tid >= total) return;
    int j    = tid & 15;
    int lane = (tid >> 4) & 31;
    int rest = tid >> 9;
    int KC = (C * 9) / 32;
    int kc = rest % KC;
    int mt = rest / KC;
    int m  = mt * 16 + (lane & 15);
    int g  = lane >> 4;
    int K  = kc * 32 + j + 16 * g;
    float val = 0.f;
    if (m < O) {
        int tap = K / C, c = K % C;
        int ki = tap / 3, kj = tap - ki * 3;
        val = w[((m * C + c) * 3 + ki) * 3 + kj];
    }
    pk[tid] = (_Float16)val;
}

// ---------------------------------------------------------------------------
// Tiny MLP: g[b][j] = relu(bsize@gw1.T+gb1) @ gw2.T + gb2   (8x32)
// ---------------------------------------------------------------------------
__global__ void mlp_g(const float* __restrict__ bs, const float* __restrict__ gw1,
                      const float* __restrict__ gb1, const float* __restrict__ gw2,
                      const float* __restrict__ gb2, float* __restrict__ g) {
    int tid = threadIdx.x;
    if (tid >= 8 * 32) return;
    int b = tid >> 5, j = tid & 31;
    float b0 = bs[b * 2], b1 = bs[b * 2 + 1];
    float acc = gb2[j];
    for (int k = 0; k < 64; ++k) {
        float hk = b0 * gw1[k * 2] + b1 * gw1[k * 2 + 1] + gb1[k];
        hk = hk > 0.f ? hk : 0.f;
        acc += hk * gw2[j * 64 + k];
    }
    g[tid] = acc;
}

// Broadcast g into channels 0..31 of the fused NHWC(64) f16 tensor.
__global__ void bcast_g(const float* __restrict__ g, _Float16* __restrict__ fused, int total) {
    int tid = blockIdx.x * blockDim.x + threadIdx.x;
    if (tid >= total) return;
    int c = tid & 31;
    int pix = tid >> 5;
    int b = pix >> 14;
    fused[pix * 64 + c] = (_Float16)g[b * 32 + c];
}

// ---------------------------------------------------------------------------
// 3x3 SAME conv as implicit GEMM with WMMA f32<-f16.
// One wave per (16 out-ch) x (16 pixels along W) tile.
// Input: NHWC f16 with CIN channels. Output: NHWC f16 (or f32 for offsets).
// ---------------------------------------------------------------------------
template <int CIN, bool RELU, bool OUTF32>
__global__ __launch_bounds__(32) void conv3x3_wmma(
    const _Float16* __restrict__ in, const _Float16* __restrict__ wpk,
    const float* __restrict__ bias, _Float16* __restrict__ outh,
    float* __restrict__ outf, int Ocount, int CstrideOut, int co_base) {
    constexpr int KC = (CIN * 9) / 32;
    int lane = threadIdx.x;
    int g = lane >> 4, n = lane & 15;
    int t  = blockIdx.x;
    int w0 = (t & 7) * 16;          // W/16 == 8
    int h  = (t >> 3) & 127;
    int b  = t >> 10;
    int mt = blockIdx.y;

    v8f acc = {};
#pragma unroll
    for (int kc = 0; kc < KC; ++kc) {
        int tap = (kc * 32) / CIN;
        int cb  = (kc * 32) % CIN;
        int ki = tap / 3, kj = tap - ki * 3;
        int y  = h + ki - 1;
        int xc = w0 + n + kj - 1;
        v16h bf = {};
        if (y >= 0 && y < HH && xc >= 0 && xc < WW) {
            bf = *(const v16h*)(in + ((b * HH + y) * WW + xc) * CIN + cb + 16 * g);
        }
        v16h af = *(const v16h*)(wpk + ((mt * KC + kc) * 32 + lane) * 16);
        acc = WMMA_F16(af, bf, acc);
    }

    int pixo = (b * HH + h) * WW + w0 + n;
    if constexpr (OUTF32) {
#pragma unroll
        for (int r = 0; r < 8; ++r) {
            int m = mt * 16 + 8 * g + r;
            if (m < Ocount) {
                float v = acc[r] + bias[m];
                if (RELU) v = v > 0.f ? v : 0.f;
                outf[pixo * CstrideOut + co_base + m] = v;
            }
        }
    } else {
        if (mt * 16 + 8 * g < Ocount) {
            v8h ov;
#pragma unroll
            for (int r = 0; r < 8; ++r) {
                float v = acc[r] + bias[mt * 16 + 8 * g + r];
                if (RELU) v = v > 0.f ? v : 0.f;
                ov[r] = (_Float16)v;
            }
            *(v8h*)(outh + pixo * CstrideOut + co_base + mt * 16 + 8 * g) = ov;
        }
    }
}

// ---------------------------------------------------------------------------
// Deformable conv 64->64, 3x3, bilinear sampling, fused bias+ReLU, NCHW f32 out.
// 128 threads: 4 waves cooperatively build sampled B fragments for all 18
// K-chunks in LDS (B-fragment layout), then each wave owns one 16-out-ch tile
// and runs 18 WMMAs reading B from LDS.
// ---------------------------------------------------------------------------
__global__ __launch_bounds__(128) void deform_wmma(
    const _Float16* __restrict__ xh, const float* __restrict__ offs,
    const _Float16* __restrict__ wpk, const float* __restrict__ bias,
    float* __restrict__ out) {
    __shared__ __align__(128) _Float16 smem[18 * 512];   // 18 KB
    int lane = threadIdx.x & 31;
    int wv   = threadIdx.x >> 5;
    int g = lane >> 4, n = lane & 15;
    int t  = blockIdx.x;
    int w0 = (t & 7) * 16;
    int h  = (t >> 3) & 127;
    int b  = t >> 10;
    int pix = (b * HH + h) * WW + w0 + n;

    // Phase 1: bilinear sampling into LDS (chunks round-robined over waves)
    for (int ch = wv; ch < 18; ch += 4) {
        int tap = ch >> 1;
        int cb  = (ch & 1) * 32 + 16 * g;
        int ki = tap / 3, kj = tap - ki * 3;
        float dy = offs[pix * 18 + 2 * tap];
        float dx = offs[pix * 18 + 2 * tap + 1];
        float py = (float)(h + ki - 1) + dy;
        float px = (float)(w0 + n + kj - 1) + dx;
        float y0f = floorf(py), x0f = floorf(px);
        float wy1 = py - y0f, wy0 = 1.f - wy1;
        float wx1 = px - x0f, wx0 = 1.f - wx1;
        float facc[16];
#pragma unroll
        for (int i = 0; i < 16; ++i) facc[i] = 0.f;
#pragma unroll
        for (int c4 = 0; c4 < 4; ++c4) {
            float ycf = y0f + (float)(c4 >> 1);
            float xcf = x0f + (float)(c4 & 1);
            float cw = ((c4 >> 1) ? wy1 : wy0) * ((c4 & 1) ? wx1 : wx0);
            if (ycf >= 0.f && ycf <= (float)(HH - 1) && xcf >= 0.f && xcf <= (float)(WW - 1)) {
                int yi = (int)ycf, xi = (int)xcf;
                v16h pv = *(const v16h*)(xh + ((b * HH + yi) * WW + xi) * 64 + cb);
#pragma unroll
                for (int i = 0; i < 16; ++i) facc[i] += cw * (float)pv[i];
            }
        }
        v16h bf;
#pragma unroll
        for (int i = 0; i < 16; ++i) bf[i] = (_Float16)facc[i];
        *(v16h*)(smem + ch * 512 + lane * 16) = bf;
    }
    __syncthreads();

    // Phase 2: 18-chunk WMMA GEMM, one out-channel tile per wave
    v8f acc = {};
#pragma unroll
    for (int ch = 0; ch < 18; ++ch) {
        v16h bf = *(const v16h*)(smem + ch * 512 + lane * 16);
        v16h af = *(const v16h*)(wpk + ((wv * 18 + ch) * 32 + lane) * 16);
        acc = WMMA_F16(af, bf, acc);
    }
#pragma unroll
    for (int r = 0; r < 8; ++r) {
        int co = wv * 16 + 8 * g + r;
        float v = acc[r] + bias[co];
        v = v > 0.f ? v : 0.f;
        out[((b * OO + co) * HH + h) * WW + w0 + n] = v;
    }
}

// ---------------------------------------------------------------------------
extern "C" void kernel_launch(void* const* d_in, const int* in_sizes, int n_in,
                              void* d_out, int out_size, void* d_ws, size_t ws_size,
                              hipStream_t stream) {
    (void)in_sizes; (void)n_in; (void)out_size; (void)ws_size;
    const float* x   = (const float*)d_in[0];
    const float* bs  = (const float*)d_in[1];
    const float* gw1 = (const float*)d_in[2];
    const float* gb1 = (const float*)d_in[3];
    const float* gw2 = (const float*)d_in[4];
    const float* gb2 = (const float*)d_in[5];
    const float* lw1 = (const float*)d_in[6];
    const float* lb1 = (const float*)d_in[7];
    const float* lw2 = (const float*)d_in[8];
    const float* lb2 = (const float*)d_in[9];
    const float* fw1 = (const float*)d_in[10];
    const float* fb1 = (const float*)d_in[11];
    const float* fw2 = (const float*)d_in[12];
    const float* fb2 = (const float*)d_in[13];
    const float* dw  = (const float*)d_in[14];
    const float* db  = (const float*)d_in[15];
    float* out = (float*)d_out;

    char* ws = (char*)d_ws;
    size_t off = 0;
    auto alloc = [&](size_t bytes) -> char* {
        char* p = ws + off;
        off = (off + bytes + 255) & ~(size_t)255;
        return p;
    };
    _Float16* xh     = (_Float16*)alloc((size_t)BB * HH * WW * 64 * 2);
    _Float16* y1     = (_Float16*)alloc((size_t)BB * HH * WW * 32 * 2);
    _Float16* fused  = (_Float16*)alloc((size_t)BB * HH * WW * 64 * 2);
    _Float16* y2     = (_Float16*)alloc((size_t)BB * HH * WW * 32 * 2);
    float*    offsb  = (float*)   alloc((size_t)BB * HH * WW * 18 * 4);
    float*    gbuf   = (float*)   alloc(8 * 32 * 4);
    _Float16* pk_lw1 = (_Float16*)alloc(2 * 18 * 512 * 2);
    _Float16* pk_lw2 = (_Float16*)alloc(2 * 9  * 512 * 2);
    _Float16* pk_fw1 = (_Float16*)alloc(2 * 18 * 512 * 2);
    _Float16* pk_fw2 = (_Float16*)alloc(2 * 9  * 512 * 2);
    _Float16* pk_dw  = (_Float16*)alloc(4 * 18 * 512 * 2);

    { int total = BB * 64 * HH * WW; to_nhwc_f16<<<(total + 255) / 256, 256, 0, stream>>>(x, xh, total); }
    { int total = 2 * 18 * 512; pack_w<<<(total + 255) / 256, 256, 0, stream>>>(lw1, pk_lw1, 32, 64, total); }
    { int total = 2 * 9  * 512; pack_w<<<(total + 255) / 256, 256, 0, stream>>>(lw2, pk_lw2, 32, 32, total); }
    { int total = 2 * 18 * 512; pack_w<<<(total + 255) / 256, 256, 0, stream>>>(fw1, pk_fw1, 32, 64, total); }
    { int total = 2 * 9  * 512; pack_w<<<(total + 255) / 256, 256, 0, stream>>>(fw2, pk_fw2, 18, 32, total); }
    { int total = 4 * 18 * 512; pack_w<<<(total + 255) / 256, 256, 0, stream>>>(dw,  pk_dw,  64, 64, total); }
    mlp_g<<<1, 256, 0, stream>>>(bs, gw1, gb1, gw2, gb2, gbuf);
    { int total = BB * HH * WW * 32; bcast_g<<<(total + 255) / 256, 256, 0, stream>>>(gbuf, fused, total); }

    dim3 cgrid(BB * HH * (WW / 16), 2);
    conv3x3_wmma<64, true,  false><<<cgrid, 32, 0, stream>>>(xh,    pk_lw1, lb1, y1,      nullptr, 32, 32, 0);
    conv3x3_wmma<32, false, false><<<cgrid, 32, 0, stream>>>(y1,    pk_lw2, lb2, fused,   nullptr, 32, 64, 32);
    conv3x3_wmma<64, true,  false><<<cgrid, 32, 0, stream>>>(fused, pk_fw1, fb1, y2,      nullptr, 32, 32, 0);
    conv3x3_wmma<32, false, true ><<<cgrid, 32, 0, stream>>>(y2,    pk_fw2, fb2, nullptr, offsb,   18, 18, 0);

    deform_wmma<<<BB * HH * (WW / 16), 128, 0, stream>>>(xh, offsb, pk_dw, db, out);
}
